// AggregatedAttention_1511828488885
// MI455X (gfx1250) — matrix operand order repaired and checked by reading, compiled, not run on hardware
//
#include <hip/hip_runtime.h>
#include <hip/hip_bf16.h>
#include <math.h>

// ---------------- problem constants ----------------
static constexpr int Bc    = 4;
static constexpr int HIMG  = 64;
static constexpr int WIMG  = 64;
static constexpr int DIM   = 128;
static constexpr int HEADS = 4;
static constexpr int HD    = 32;                 // DIM / HEADS
static constexpr int Nn    = HIMG * WIMG;        // 4096
static constexpr int SR    = 4;
static constexpr int PW    = WIMG / SR;          // 16
static constexpr int PL    = (HIMG / SR) * PW;   // 256
static constexpr int LL    = 9;
static constexpr int TBL   = 2048;
static constexpr int CCOLS = 512;                // q(128) | kv(256) | sr(128)

typedef __attribute__((ext_vector_type(16))) __bf16 v16bf;
typedef __attribute__((ext_vector_type(8)))  float  v8f;

union Frag16 { v16bf v; unsigned short u[16]; };

#if __has_builtin(__builtin_amdgcn_tensor_load_to_lds)
#define HAVE_TDM 1
typedef __attribute__((ext_vector_type(4))) unsigned int u32x4;
typedef __attribute__((ext_vector_type(8))) int          i32x8;
typedef __attribute__((ext_vector_type(4))) int          i32x4;
#else
#define HAVE_TDM 0
#endif

__device__ __forceinline__ unsigned short f2bf(float f) {
  union { float f; unsigned u; } x; x.f = f;
  unsigned u = x.u;
  u += 0x7FFFu + ((u >> 16) & 1u);               // round-to-nearest-even
  return (unsigned short)(u >> 16);
}

__device__ __forceinline__ v8f wmma_bf16(const Frag16 &a, const Frag16 &b, v8f c) {
  // emits v_wmma_f32_16x16x32_bf16
  return __builtin_amdgcn_wmma_f32_16x16x32_bf16(false, a.v, false, b.v,
                                                 (short)0, c, false, false);
}

// A-frag (16-bit A 16x32, ISA 7.12.2): element j of lane-half lh holds K = lh*8 + j + 8*(j>=8)
__device__ __forceinline__ int a_k(int j, int lh) { return lh * 8 + j + ((j >= 8) ? 8 : 0); }

__device__ __forceinline__ float gelu_exact(float x) {
  return 0.5f * x * (1.0f + erff(x * 0.70710678118654752f));
}

// async global -> LDS copy of one 16-byte chunk (CDNA5, ASYNCcnt-tracked)
__device__ __forceinline__ void async_copy_b128(void* lds_dst, const void* gsrc) {
  unsigned ldst = (unsigned)(size_t)lds_dst;     // low 32 bits of generic LDS ptr = LDS addr
  unsigned long long ga = (unsigned long long)(size_t)gsrc;
  asm volatile("global_load_async_to_lds_b128 %0, %1, off"
               :: "v"(ldst), "v"(ga) : "memory");
}
__device__ __forceinline__ void async_wait0() {
  asm volatile("s_wait_asynccnt 0x0" ::: "memory");
}

// =====================================================================
// K0: weight prep.  WcatT[c][k] (column-major bf16, c in [0,512)),
//     WprojT[c][k], bcat[512] = bq|bkv|bsr
// =====================================================================
__global__ void k_prep(const float* __restrict__ Wq, const float* __restrict__ Wkv,
                       const float* __restrict__ Wsr, const float* __restrict__ Wproj,
                       const float* __restrict__ bq, const float* __restrict__ bkv,
                       const float* __restrict__ bsr,
                       unsigned short* __restrict__ WcatT,
                       unsigned short* __restrict__ WprojT,
                       float* __restrict__ bcat) {
  int idx = blockIdx.x * blockDim.x + threadIdx.x;
  if (idx < DIM * CCOLS) {
    int k = idx / CCOLS, c = idx % CCOLS;
    float w;
    if (c < 128)      w = Wq[k * 128 + c];
    else if (c < 384) w = Wkv[k * 256 + (c - 128)];
    else              w = Wsr[k * 128 + (c - 384)];
    WcatT[(size_t)c * DIM + k] = f2bf(w);
  }
  int i2 = idx - DIM * CCOLS;
  if (i2 >= 0 && i2 < DIM * DIM) {
    int k = i2 / DIM, c = i2 % DIM;
    WprojT[(size_t)c * DIM + k] = f2bf(Wproj[k * DIM + c]);
  }
  if (idx < CCOLS) {
    bcat[idx] = (idx < 128) ? bq[idx] : (idx < 384 ? bkv[idx - 128] : bsr[idx - 384]);
  }
}

// =====================================================================
// Generic WMMA GEMM:  C[M,Ncols] = A[M,128] @ W + bias, gelu on cols>=gelu_from
// W column-major bf16: W[c*128 + k].  One 16-row tile per block (M must be
// a multiple of 16, A row stride = 128 contiguous), each wave covers 64
// output columns (4 N-tiles), K-loop = 4 x 32.
// A tile staged fp32 via async global->LDS DMA, converted to bf16 in LDS.
// =====================================================================
__global__ void __launch_bounds__(256)
k_gemm_bf16(const float* __restrict__ A,
            const unsigned short* __restrict__ Wt,
            const float* __restrict__ bias,
            float* __restrict__ C, int ldc,
            int M, int Ncols, int gelu_from) {
  constexpr int K = 128;
  __shared__ __align__(16) float          ldsAf[16 * K];   // raw fp32 tile (8 KB)
  __shared__ __align__(16) unsigned short ldsA [16 * K];   // bf16 tile (4 KB)

  const int m0 = blockIdx.x * 16;
  const int tid = threadIdx.x, nthr = blockDim.x;

  // async DMA the 8KB fp32 tile into LDS (16*128 floats = 512 x b128 chunks)
  const float* tileSrc = A + (size_t)m0 * K;
  for (int c = tid; c < (16 * K) / 4; c += nthr)
    async_copy_b128(&ldsAf[c * 4], &tileSrc[c * 4]);
  async_wait0();
  __syncthreads();
  for (int e = tid; e < 16 * K; e += nthr) ldsA[e] = f2bf(ldsAf[e]);
  __syncthreads();

  const int wave = tid >> 5, lane = tid & 31;
  const int lh = lane >> 4, r = lane & 15;
  const int waves = nthr >> 5;
  const int colBase = (blockIdx.y * waves + wave) * 64;
  if (colBase < Ncols)
    __builtin_prefetch(Wt + (size_t)colBase * K, 0, 1);   // global_prefetch

  Frag16 af[4];
  #pragma unroll
  for (int ks = 0; ks < 4; ++ks) {
    const int k0 = ks * 32;
    #pragma unroll
    for (int j = 0; j < 16; ++j) af[ks].u[j] = ldsA[r * K + k0 + a_k(j, lh)];
  }

  for (int nt = 0; nt < 4; ++nt) {
    const int n0 = colBase + nt * 16;
    if (n0 >= Ncols) break;                 // wave-uniform
    v8f acc = {};
    #pragma unroll
    for (int ks = 0; ks < 4; ++ks) {
      const int k0 = ks * 32;
      Frag16 bf_;
      #pragma unroll
      for (int j = 0; j < 16; ++j)
        bf_.u[j] = Wt[(size_t)(n0 + r) * K + k0 + a_k(j, lh)];
      acc = wmma_bf16(af[ks], bf_, acc);
    }
    #pragma unroll
    for (int i = 0; i < 8; ++i) {
      int row = m0 + i + lh * 8;            // C: M = i + 8*half, N = lane&15
      int col = n0 + r;
      if (row < M) {
        float v = acc[i] + bias[col];
        if (col >= gelu_from) v = gelu_exact(v);
        C[(size_t)row * ldc + col] = v;
      }
    }
  }
}

// =====================================================================
// K2: q/k/v epilogue over (b,n,h)
// =====================================================================
__global__ void k_qkv_epilogue(const float* __restrict__ C1,
                               const float* __restrict__ qemb,
                               const float* __restrict__ temp,
                               const float* __restrict__ sls,
                               unsigned short* __restrict__ qs_bf,
                               float* __restrict__ qs_f,
                               float* __restrict__ qn_f,
                               float* __restrict__ kn_f,
                               float* __restrict__ v_f) {
  int idx = blockIdx.x * blockDim.x + threadIdx.x;
  if (idx >= Bc * Nn * HEADS) return;
  int h = idx % HEADS, bn = idx / HEADS;
  int n = bn % Nn, b = bn / Nn;
  const float* row = C1 + (size_t)bn * CCOLS;

  // ---- q ----
  const float* q = row + h * HD;
  float ss = 0.f;
  #pragma unroll
  for (int d = 0; d < HD; ++d) ss += q[d] * q[d];
  float inv = 1.0f / fmaxf(sqrtf(ss), 1e-12f);
  float scale = log1pf(expf(temp[h])) * sls[0];
  size_t qb = ((size_t)(b * HEADS + h) * Nn + n) * HD;
  #pragma unroll
  for (int d = 0; d < HD; ++d) {
    float qn = q[d] * inv;
    qn_f[qb + d] = qn;
    float qs = (qn + qemb[h * HD + d]) * scale;
    qs_f[qb + d] = qs;
    qs_bf[qb + d] = f2bf(qs);
  }
  // ---- k (l2norm) ----
  const float* k = row + 128 + h * HD;
  ss = 0.f;
  #pragma unroll
  for (int d = 0; d < HD; ++d) ss += k[d] * k[d];
  inv = 1.0f / fmaxf(sqrtf(ss), 1e-12f);
  size_t kb = ((size_t)bn * HEADS + h) * HD;   // (b, pixel, head, d) for 3x3 window reads
  #pragma unroll
  for (int d = 0; d < HD; ++d) kn_f[kb + d] = k[d] * inv;
  // ---- v ----
  const float* vv = row + 256 + h * HD;
  #pragma unroll
  for (int d = 0; d < HD; ++d) v_f[kb + d] = vv[d];
}

// =====================================================================
// K3: 4x4 mean-pool of gelu(sr) + layernorm -> y[B*PL,128]
// =====================================================================
__global__ void __launch_bounds__(128)
k_pool_ln(const float* __restrict__ C1, const float* __restrict__ ln_g,
          const float* __restrict__ ln_b, float* __restrict__ y) {
  __shared__ float red[128];
  int bp = blockIdx.x;            // b*PL + p
  int b = bp / PL, p = bp % PL;
  int ph = p / PW, pw = p % PW;
  int d = threadIdx.x;

  float s = 0.f;
  for (int ii = 0; ii < SR; ++ii)
    for (int jj = 0; jj < SR; ++jj) {
      int n = (ph * SR + ii) * WIMG + (pw * SR + jj);
      s += C1[((size_t)b * Nn + n) * CCOLS + 384 + d];
    }
  float val = s * (1.0f / 16.0f);

  red[d] = val; __syncthreads();
  for (int st = 64; st > 0; st >>= 1) { if (d < st) red[d] += red[d + st]; __syncthreads(); }
  float mean = red[0] * (1.0f / 128.0f); __syncthreads();
  float df = val - mean;
  red[d] = df * df; __syncthreads();
  for (int st = 64; st > 0; st >>= 1) { if (d < st) red[d] += red[d + st]; __syncthreads(); }
  float var = red[0] * (1.0f / 128.0f);
  y[(size_t)bp * DIM + d] = df * rsqrtf(var + 1e-5f) * ln_g[d] + ln_b[d];
}

// =====================================================================
// K5: kvp epilogue: k_pool l2norm -> bf16 (b,h,p,d); v_pool -> bf16 (b,h,d,p)
// =====================================================================
__global__ void k_poolkv_epilogue(const float* __restrict__ kvp,
                                  unsigned short* __restrict__ kpool_bf,
                                  unsigned short* __restrict__ vpoolT_bf) {
  int idx = blockIdx.x * blockDim.x + threadIdx.x;
  if (idx >= Bc * HEADS * PL) return;
  int p = idx % PL, bh = idx / PL;
  int h = bh % HEADS, b = bh / HEADS;
  const float* row = kvp + (size_t)(b * PL + p) * 256;
  const float* k = row + h * HD;
  const float* v = row + 128 + h * HD;
  float ss = 0.f;
  #pragma unroll
  for (int d = 0; d < HD; ++d) ss += k[d] * k[d];
  float inv = 1.0f / fmaxf(sqrtf(ss), 1e-12f);
  #pragma unroll
  for (int d = 0; d < HD; ++d) {
    kpool_bf[((size_t)bh * PL + p) * HD + d] = f2bf(k[d] * inv);
    vpoolT_bf[((size_t)bh * HD + d) * PL + p] = f2bf(v[d]);
  }
}

// =====================================================================
// K6: cpb MLP: relu(table @ w1 + b1) @ w2 + b2 -> cpb[TBL][HEADS]
// =====================================================================
__global__ void k_cpb(const float* __restrict__ tbl, const float* __restrict__ w1,
                      const float* __restrict__ b1, const float* __restrict__ w2,
                      const float* __restrict__ b2, float* __restrict__ cpb) {
  int t = blockIdx.x * blockDim.x + threadIdx.x;
  if (t >= TBL) return;
  float c0 = tbl[t * 2 + 0], c1 = tbl[t * 2 + 1];
  float acc0 = b2[0], acc1 = b2[1], acc2 = b2[2], acc3 = b2[3];
  for (int j = 0; j < 512; ++j) {
    float hv = fmaxf(c0 * w1[j] + c1 * w1[512 + j] + b1[j], 0.0f);
    acc0 += hv * w2[j * HEADS + 0];
    acc1 += hv * w2[j * HEADS + 1];
    acc2 += hv * w2[j * HEADS + 2];
    acc3 += hv * w2[j * HEADS + 3];
  }
  cpb[t * HEADS + 0] = acc0; cpb[t * HEADS + 1] = acc1;
  cpb[t * HEADS + 2] = acc2; cpb[t * HEADS + 3] = acc3;
}

// =====================================================================
// K7: attention.  2 waves/block, each wave: 16 query rows of one (b,h).
//   k_pool tile (256x32 bf16, 16KB) DMA'd to LDS once per block via TDM.
//   pool logits: 16 WMMAs; local logits scalar; softmax over 265; PV: 16 WMMAs.
// =====================================================================
__global__ void __launch_bounds__(64)
k_attn(const unsigned short* __restrict__ qs_bf, const float* __restrict__ qs_f,
       const float* __restrict__ qn_f, const float* __restrict__ kn_f,
       const float* __restrict__ v_f, const unsigned short* __restrict__ kpool_bf,
       const unsigned short* __restrict__ vpoolT_bf, const float* __restrict__ cpb,
       const int* __restrict__ rpi, const float* __restrict__ rbl,
       const float* __restrict__ ltok, const float* __restrict__ lbias,
       float* __restrict__ out_pre) {
  constexpr int AS = 272;                     // attn row stride (9 local + 256 pool, padded)
  __shared__ float attnS[2 * 16 * AS];
  __shared__ float wlS[2 * 16 * 12];
  __shared__ float outS[2 * 16 * 36];
  __shared__ __align__(16) unsigned short kpLDS[PL * HD];   // 16 KB k_pool tile

  const int wave = threadIdx.x >> 5, lane = threadIdx.x & 31;
  const int lh = lane >> 4, r = lane & 15;
  const int tilesPerPlane = Nn / 32;
  const int bh = blockIdx.x / tilesPerPlane;
  const int tile = blockIdx.x % tilesPerPlane;
  const int h = bh % HEADS, b = bh / HEADS;
  const int m0 = tile * 32 + wave * 16;

  float* attn = attnS + wave * 16 * AS;
  float* wl   = wlS   + wave * 16 * 12;
  float* outT = outS  + wave * 16 * 36;

  // ---- TDM: DMA the (b,h) k_pool plane into LDS (2D: 256 rows x 32 bf16) ----
  const unsigned short* kpG = kpool_bf + (size_t)bh * PL * HD;
#if HAVE_TDM
  if (wave == 0) {
    unsigned long long ga = (unsigned long long)(size_t)kpG;
    unsigned lds_addr = (unsigned)(size_t)(void*)kpLDS;
    u32x4 g0;
    g0.x = 0x1u;                                            // count=1 (user descriptor)
    g0.y = lds_addr;                                        // lds_addr [63:32]
    g0.z = (unsigned)(ga & 0xFFFFFFFFu);                    // global_addr [95:64]
    g0.w = (unsigned)((ga >> 32) & 0x01FFFFFFu) | (2u << 30); // addr hi + type=2
    i32x8 g1;
    g1[0] = (int)(1u << 16);                    // workgroup_mask=0, data_size=1 (2B)
    g1[1] = (int)((unsigned)HD << 16);          // tensor_dim0 low16 @ [63:48]
    g1[2] = (int)((unsigned)PL << 16);          // tensor_dim1 low16 @ [95:80]
    g1[3] = (int)((unsigned)HD << 16);          // tile_dim0 @ [127:112]
    g1[4] = (int)PL;                            // tile_dim1 @ [143:128]
    g1[5] = (int)HD;                            // tensor_dim0_stride low32
    g1[6] = 0;
    g1[7] = 0;
    i32x4 gz = {0, 0, 0, 0};
#if __clang_major__ >= 23
    i32x8 gz8 = {0, 0, 0, 0, 0, 0, 0, 0};
    __builtin_amdgcn_tensor_load_to_lds(g0, g1, gz, gz, gz8, 0);
#else
    __builtin_amdgcn_tensor_load_to_lds(g0, g1, gz, gz, 0);
#endif
    __builtin_amdgcn_s_wait_tensorcnt(0);
  }
#else
  for (int e = threadIdx.x; e < PL * HD; e += 64) kpLDS[e] = kpG[e];
#endif
  __syncthreads();

  const size_t qbase = ((size_t)bh * Nn + m0) * HD;

  // ---- q_scaled A-fragment (16x32) ----
  Frag16 qa;
  #pragma unroll
  for (int j = 0; j < 16; ++j)
    qa.u[j] = qs_bf[qbase + (size_t)r * HD + a_k(j, lh)];

  // ---- pool logits: q @ k_pool^T, 16 tiles of 16 (B-frags from LDS) ----
  for (int nt = 0; nt < PL / 16; ++nt) {
    const int p0 = nt * 16;
    Frag16 kb;
    #pragma unroll
    for (int j = 0; j < 16; ++j)
      kb.u[j] = kpLDS[(p0 + r) * HD + a_k(j, lh)];
    v8f acc = {};
    acc = wmma_bf16(qa, kb, acc);
    #pragma unroll
    for (int i = 0; i < 8; ++i) {
      int row = i + lh * 8;
      int n = m0 + row;
      int p = p0 + r;
      float bias = cpb[rpi[(size_t)n * PL + p] * HEADS + h];
      attn[row * AS + LL + p] = acc[i] + bias;
    }
  }

  // ---- local logits + learnable-token weights (lane = row r, half splits the 9 taps) ----
  {
    const int n = m0 + r;
    const int pi = n / WIMG, pj = n % WIMG;
    const float* qrow  = qs_f + qbase + (size_t)r * HD;
    const float* qnrow = qn_f + qbase + (size_t)r * HD;
    const int l0 = lh ? 5 : 0, l1 = lh ? 9 : 5;
    for (int l = l0; l < l1; ++l) {
      int ni = pi + (l / 3) - 1, nj = pj + (l % 3) - 1;
      float a;
      if (ni < 0 || ni >= HIMG || nj < 0 || nj >= WIMG) {
        a = -__builtin_inff();                 // analytic border mask
      } else {
        const float* krow = kn_f + (((size_t)b * Nn + ni * WIMG + nj) * HEADS + h) * HD;
        float s = 0.f;
        #pragma unroll
        for (int d = 0; d < HD; ++d) s += qrow[d] * krow[d];
        a = s + rbl[h * LL + l];
      }
      attn[r * AS + l] = a;
      float ss = 0.f;
      #pragma unroll
      for (int d = 0; d < HD; ++d) ss += qnrow[d] * ltok[(h * HD + d) * LL + l];
      wl[r * 12 + l] = ss + lbias[h * LL + l];
    }
  }
  __syncthreads();

  // ---- softmax over 265 logits (lanes 0..15, one row each) ----
  if (lh == 0) {
    float* arow = attn + r * AS;
    float mx = -__builtin_inff();
    for (int c = 0; c < LL + PL; ++c) mx = fmaxf(mx, arow[c]);
    float s = 0.f;
    for (int c = 0; c < LL + PL; ++c) { float e = __expf(arow[c] - mx); arow[c] = e; s += e; }
    float invs = 1.0f / s;
    for (int c = 0; c < LL + PL; ++c) arow[c] *= invs;
  }
  __syncthreads();

  // ---- x_pool = a_pool @ v_pool  (2 N-tiles x 8 K-steps) ----
  const unsigned short* vp = vpoolT_bf + (size_t)bh * HD * PL;
  for (int ntile = 0; ntile < 2; ++ntile) {
    const int nb = ntile * 16;
    v8f acc = {};
    for (int ks = 0; ks < 8; ++ks) {
      const int k0 = ks * 32;
      Frag16 aa, vb;
      #pragma unroll
      for (int j = 0; j < 16; ++j)
        aa.u[j] = f2bf(attn[r * AS + LL + k0 + a_k(j, lh)]);
      #pragma unroll
      for (int j = 0; j < 16; ++j)
        vb.u[j] = vp[(size_t)(nb + r) * PL + k0 + a_k(j, lh)];
      acc = wmma_bf16(aa, vb, acc);
    }
    #pragma unroll
    for (int i = 0; i < 8; ++i)
      outT[(i + lh * 8) * 36 + nb + r] = acc[i];
  }
  __syncthreads();

  // ---- x_local: lane (r, lh) owns row r, dims [lh*16, lh*16+16) ----
  {
    const int n = m0 + r;
    const int pi = n / WIMG, pj = n % WIMG;
    const int d0 = lh * 16;
    float accv[16];
    #pragma unroll
    for (int d = 0; d < 16; ++d) accv[d] = 0.f;
    for (int l = 0; l < LL; ++l) {
      int ni = pi + (l / 3) - 1, nj = pj + (l % 3) - 1;
      if (ni < 0 || ni >= HIMG || nj < 0 || nj >= WIMG) continue;  // padded v == 0
      float w = wl[r * 12 + l] + attn[r * AS + l];
      const float* vrow = v_f + (((size_t)b * Nn + ni * WIMG + nj) * HEADS + h) * HD + d0;
      #pragma unroll
      for (int d = 0; d < 16; ++d) accv[d] += w * vrow[d];
    }
    #pragma unroll
    for (int d = 0; d < 16; ++d) outT[r * 36 + d0 + d] += accv[d];
  }
  __syncthreads();

  // ---- store out_pre (b, n, h, d) ----
  for (int t = lane; t < 16 * 32; t += 32) {
    int row = t >> 5, d = t & 31;
    out_pre[(((size_t)b * Nn + m0 + row) * HEADS + h) * HD + d] = outT[row * 36 + d];
  }
}

// =====================================================================
// launcher
// =====================================================================
extern "C" void kernel_launch(void* const* d_in, const int* in_sizes, int n_in,
                              void* d_out, int out_size, void* d_ws, size_t ws_size,
                              hipStream_t stream) {
  const float* x     = (const float*)d_in[0];
  const float* rct   = (const float*)d_in[1];
  const float* sls   = (const float*)d_in[2];
  const float* Wq    = (const float*)d_in[3];
  const float* bq    = (const float*)d_in[4];
  const float* Wkv   = (const float*)d_in[5];
  const float* bkv   = (const float*)d_in[6];
  const float* temp  = (const float*)d_in[7];
  const float* qemb  = (const float*)d_in[8];
  const float* Wsr   = (const float*)d_in[9];
  const float* bsr   = (const float*)d_in[10];
  const float* ln_g  = (const float*)d_in[11];
  const float* ln_b  = (const float*)d_in[12];
  const float* cw1   = (const float*)d_in[13];
  const float* cb1   = (const float*)d_in[14];
  const float* cw2   = (const float*)d_in[15];
  const float* cb2   = (const float*)d_in[16];
  const float* rbl   = (const float*)d_in[17];
  const float* ltok  = (const float*)d_in[18];
  const float* lbias = (const float*)d_in[19];
  const float* Wproj = (const float*)d_in[20];
  const float* bproj = (const float*)d_in[21];
  const int*   rpi   = (const int*)d_in[24];
  // d_in[22]=H, d_in[23]=W hardcoded (64); d_in[25] padding_mask computed analytically

  char* ws = (char*)d_ws;
  size_t off = 0;
  auto alloc = [&](size_t bytes) {
    void* p = ws + off;
    off = (off + bytes + 255) & ~(size_t)255;
    return p;
  };
  unsigned short* WcatT   = (unsigned short*)alloc((size_t)DIM * CCOLS * 2);
  unsigned short* WprojT  = (unsigned short*)alloc((size_t)DIM * DIM * 2);
  float*          bcat    = (float*)alloc(CCOLS * 4);
  float*          C1      = (float*)alloc((size_t)Bc * Nn * CCOLS * 4);
  unsigned short* qs_bf   = (unsigned short*)alloc((size_t)Bc * Nn * DIM * 2);
  float*          qs_f    = (float*)alloc((size_t)Bc * Nn * DIM * 4);
  float*          qn_f    = (float*)alloc((size_t)Bc * Nn * DIM * 4);
  float*          kn_f    = (float*)alloc((size_t)Bc * Nn * DIM * 4);
  float*          v_f     = (float*)alloc((size_t)Bc * Nn * DIM * 4);
  float*          y       = (float*)alloc((size_t)Bc * PL * DIM * 4);
  float*          kvp     = (float*)alloc((size_t)Bc * PL * 256 * 4);
  unsigned short* kpoolbf = (unsigned short*)alloc((size_t)Bc * HEADS * PL * HD * 2);
  unsigned short* vpoolbf = (unsigned short*)alloc((size_t)Bc * HEADS * PL * HD * 2);
  float*          cpbBuf  = (float*)alloc((size_t)TBL * HEADS * 4);
  float*          out_pre = (float*)alloc((size_t)Bc * Nn * DIM * 4);

  const int M1 = Bc * Nn;                         // 16384

  // K0: weight prep
  k_prep<<<dim3((DIM * CCOLS + DIM * DIM + 255) / 256), 256, 0, stream>>>(
      Wq, Wkv, Wsr, Wproj, bq, bkv, bsr, WcatT, WprojT, bcat);

  // K1: fused q|kv|sr GEMM (gelu on sr segment, cols >= 384)
  k_gemm_bf16<<<dim3(M1 / 16, 1), 256, 0, stream>>>(
      x, WcatT, bcat, C1, CCOLS, M1, CCOLS, 384);

  // K2: q/k/v epilogue
  k_qkv_epilogue<<<dim3((Bc * Nn * HEADS + 255) / 256), 256, 0, stream>>>(
      C1, qemb, temp, sls, qs_bf, qs_f, qn_f, kn_f, v_f);

  // K3: pool + layernorm
  k_pool_ln<<<dim3(Bc * PL), 128, 0, stream>>>(C1, ln_g, ln_b, y);

  // K4: kvp = y @ Wkv + bkv  (Wkv = cols 128..383 of WcatT)
  k_gemm_bf16<<<dim3(Bc * PL / 16, 1), 128, 0, stream>>>(
      y, WcatT + (size_t)128 * DIM, bkv, kvp, 256, Bc * PL, 256, 1 << 30);

  // K5: pooled k/v epilogue
  k_poolkv_epilogue<<<dim3((Bc * HEADS * PL + 255) / 256), 256, 0, stream>>>(
      kvp, kpoolbf, vpoolbf);

  // K6: cpb MLP
  k_cpb<<<dim3((TBL + 255) / 256), 256, 0, stream>>>(rct, cw1, cb1, cw2, cb2, cpbBuf);

  // K7: attention (2 waves/block, 32 query rows per block, TDM k_pool tile)
  k_attn<<<dim3(Bc * HEADS * (Nn / 32)), 64, 0, stream>>>(
      qs_bf, qs_f, qn_f, kn_f, v_f, kpoolbf, vpoolbf, cpbBuf, rpi,
      rbl, ltok, lbias, out_pre);

  // K8: final projection -> d_out
  k_gemm_bf16<<<dim3(M1 / 16, 1), 64, 0, stream>>>(
      out_pre, WprojT, bproj, (float*)d_out, DIM, M1, DIM, 1 << 30);

  (void)in_sizes; (void)n_in; (void)out_size; (void)ws_size;
}
